// WaveletBlock1d_1288490189219
// MI455X (gfx1250) — compile-verified, matrix operand order checked by reading, not run on hardware
//
#include <hip/hip_runtime.h>
#include <hip/hip_bf16.h>
#include <cstddef>

// ---------------------------------------------------------------------------
// MI455X (gfx1250) WaveletBlock1d.
// Bandwidth-bound (~1.5 GB HBM traffic @ 23.3 TB/s ~ 65us; 140 GFLOP is far
// below the v_wmma_f32_16x16x32_bf16 ceiling). Strategy:
//  * all GEMMs (conv3-as-3-shifted-GEMMs, per-scale pointwise, MLP) via
//    WMMA bf16 -> f32 accum, activations in transposed (B,N,C) layout.
//  * K dimension of every bf16 buffer is stored "fragment-swizzled" (bits 3/4
//    of the K index swapped) so each lane's 16 WMMA fragment values are one
//    contiguous 32-byte v16bf load (2 adjacent global_load_b128, no shuffles).
//  * OOB GEMM rows -> zeroed scratch row (no EXEC divergence in the K-loop),
//    4 independent accumulators/wave, K-loop unrolled x2 with two structural
//    buffer sets (no register-rotation moves, no WMMA hazard NOPs).
//  * fused epilogues: bias, beta-accumulate (conv taps), exact GELU, residual.
// ---------------------------------------------------------------------------

typedef __bf16 bf16;
typedef __bf16 v16bf __attribute__((ext_vector_type(16)));
typedef float  v8f   __attribute__((ext_vector_type(8)));

struct Filt { float lo[8]; float hi[8]; };

// Fragment swizzle: swap bits 3 and 4 of the K index (involution, per-32-block)
__device__ __forceinline__ int    swzk(int k)    { return (k & ~24) | ((k & 8) << 1) | ((k & 16) >> 1); }
__device__ __forceinline__ size_t swzks(size_t k){ return (k & ~(size_t)24) | ((k & 8) << 1) | ((k & 16) >> 1); }

__device__ __forceinline__ float geluf(float x) {
  return 0.5f * x * (1.0f + erff(x * 0.70710678118654752f));
}

#define WMMA_BF16(av, bv, cv) \
  __builtin_amdgcn_wmma_f32_16x16x32_bf16(false, (av), false, (bv), (short)0, (cv), false, false)

// ---------------------------------------------------------------------------
// Generic WMMA GEMM:  Y[b, m, n] = act( sum_k A[m,k] * X[b, n+shift, k] + ... )
// A: MxK row-major bf16 (K-swizzled).  X: (B, Ncols, K) bf16 (K-swizzled).
// Block = 256 threads = 8 waves, 4(M) x 2(N); block tile 64(M) x 128(N).
// Wave tile 16(M) x 64(N) = 4 f32 accumulators sharing one A fragment.
// K must be a multiple of 64 (here 512 or 1024).
// mode 0/1: f32 out in (B,Ncols,M) t-layout, beta=1 accumulates (conv taps)
// mode 2  : gelu -> bf16 out in t-layout, K-swizzled (MLP hidden)
// mode 3  : residual + bias -> d_out in (B, M, Ncols) final layout
// ---------------------------------------------------------------------------
__global__ __launch_bounds__(256) void wmma_gemm_kernel(
    const bf16* __restrict__ A, const bf16* __restrict__ X,
    const bf16* __restrict__ zrow,
    const float* __restrict__ bias,
    float* __restrict__ Yf, bf16* __restrict__ Yb,
    const float* __restrict__ Res,
    int M, int K, int Ncols, int shift, int mode, int beta)
{
  const int b    = blockIdx.z;
  const int tid  = threadIdx.x;
  const int lane = tid & 31;
  const int w    = tid >> 5;            // 0..7
  const int mt   = blockIdx.y * 64 + (w & 3) * 16;
  const int n0   = blockIdx.x * 128 + (w >> 2) * 64;

  const int lrow = lane & 15;
  const int koff = (lane >> 4) << 4;    // swizzled storage: upper lanes take slots 16..31

  const size_t xbatch = (size_t)b * Ncols * K;
  const bf16* Ap = A + (size_t)(mt + lrow) * K + koff;

  // Per-tile X row pointers; OOB rows -> zero row (uniform, loop-invariant).
  const bf16* Xq[4];
  #pragma unroll
  for (int j = 0; j < 4; ++j) {
    const int xr = n0 + j * 16 + lrow + shift;
    Xq[j] = ((xr >= 0 && xr < Ncols) ? (X + xbatch + (size_t)xr * K) : zrow) + koff;
  }

  v8f acc[4];
  #pragma unroll
  for (int j = 0; j < 4; ++j) acc[j] = (v8f){};

  // Two structural fragment buffers; K/64 double-steps, loads pipelined.
  v16bf a0 = *(const v16bf*)(Ap);
  v16bf a1 = *(const v16bf*)(Ap + 32);
  v16bf x0[4], x1[4];
  #pragma unroll
  for (int j = 0; j < 4; ++j) x0[j] = *(const v16bf*)(Xq[j]);
  #pragma unroll
  for (int j = 0; j < 4; ++j) x1[j] = *(const v16bf*)(Xq[j] + 32);

  const int steps = K >> 6;
  for (int s = 0; s < steps - 1; ++s) {
    const int kc = (s << 6) + 64;
    __builtin_prefetch(Ap + kc + 64, 0, 3);      // global_prefetch_b8
    __builtin_prefetch(Xq[0] + kc + 64, 0, 1);
    #pragma unroll
    for (int j = 0; j < 4; ++j) acc[j] = WMMA_BF16(a0, x0[j], acc[j]);
    a0 = *(const v16bf*)(Ap + kc);
    #pragma unroll
    for (int j = 0; j < 4; ++j) x0[j] = *(const v16bf*)(Xq[j] + kc);
    #pragma unroll
    for (int j = 0; j < 4; ++j) acc[j] = WMMA_BF16(a1, x1[j], acc[j]);
    a1 = *(const v16bf*)(Ap + kc + 32);
    #pragma unroll
    for (int j = 0; j < 4; ++j) x1[j] = *(const v16bf*)(Xq[j] + kc + 32);
  }
  #pragma unroll
  for (int j = 0; j < 4; ++j) acc[j] = WMMA_BF16(a0, x0[j], acc[j]);
  #pragma unroll
  for (int j = 0; j < 4; ++j) acc[j] = WMMA_BF16(a1, x1[j], acc[j]);

  // C/D layout (ISA 7.12.2): VGPR r, lanes<16 -> M=r, lanes>=16 -> M=8+r.
  const int row0 = mt + (lane >> 4) * 8;
  #pragma unroll
  for (int j = 0; j < 4; ++j) {
    const int col = n0 + j * 16 + lrow;
    if (col >= Ncols) continue;
    if (mode <= 1) {
      float* p = Yf + (size_t)b * Ncols * M + (size_t)col * M + row0;
      #pragma unroll
      for (int r = 0; r < 8; ++r)
        p[r] = acc[j][r] + (beta ? p[r] : bias[row0 + r]);
    } else if (mode == 2) {
      // next GEMM reads this as K: store at swizzled slots (8-run stays contiguous)
      bf16* p = Yb + (size_t)b * Ncols * M + (size_t)col * M + swzk(row0);
      #pragma unroll
      for (int r = 0; r < 8; ++r)
        p[r] = (bf16)geluf(acc[j][r] + bias[row0 + r]);
    } else {
      const float* rp = Res + (size_t)b * Ncols * M + (size_t)col * M + row0;
      #pragma unroll
      for (int r = 0; r < 8; ++r)
        Yf[((size_t)b * M + row0 + r) * Ncols + col] = rp[r] + acc[j][r] + bias[row0 + r];
    }
  }
}

// --------------------------- stats / normalization -------------------------
__global__ __launch_bounds__(256) void rowstat_kernel(
    const float* __restrict__ x, float* __restrict__ mean, float* __restrict__ rstd, int Nn)
{
  __shared__ float sS[256], sQ[256];
  const int row = blockIdx.x;
  const float* p = x + (size_t)row * Nn;
  float s = 0.f, q = 0.f;
  for (int i = threadIdx.x; i < Nn; i += 256) { float v = p[i]; s += v; q += v * v; }
  sS[threadIdx.x] = s; sQ[threadIdx.x] = q; __syncthreads();
  for (int st = 128; st > 0; st >>= 1) {
    if (threadIdx.x < st) { sS[threadIdx.x] += sS[threadIdx.x + st];
                            sQ[threadIdx.x] += sQ[threadIdx.x + st]; }
    __syncthreads();
  }
  if (threadIdx.x == 0) {
    float m = sS[0] / Nn;
    float v = sQ[0] / Nn - m * m;
    mean[row] = m;
    rstd[row] = rsqrtf((v > 0.f ? v : 0.f) + 1e-5f);
  }
}

// (B,C,N) -> normalized transposed (B,N,C) f32 + swizzled bf16, plus raw x_t.
__global__ __launch_bounds__(256) void normT_kernel(
    const float* __restrict__ x, const float* __restrict__ mean, const float* __restrict__ rstd,
    float* __restrict__ x_t, float* __restrict__ xn_t, bf16* __restrict__ xn_b,
    int Cc, int Nn)
{
  __shared__ float t0[32][33], t1[32][33];
  const int b = blockIdx.z, c0 = blockIdx.y * 32, n0 = blockIdx.x * 32;
  const int tx = threadIdx.x, ty = threadIdx.y;
  #pragma unroll
  for (int j = 0; j < 4; ++j) {
    const int c = c0 + ty + 8 * j;
    const float v = x[((size_t)b * Cc + c) * Nn + n0 + tx];
    t0[ty + 8 * j][tx] = v;
    t1[ty + 8 * j][tx] = (v - mean[b * Cc + c]) * rstd[b * Cc + c];
  }
  __syncthreads();
  #pragma unroll
  for (int j = 0; j < 4; ++j) {
    const int n = n0 + ty + 8 * j;
    const size_t idx = ((size_t)b * Nn + n) * Cc + c0 + tx;
    x_t[idx] = t0[tx][ty + 8 * j];
    const float v = t1[tx][ty + 8 * j];
    xn_t[idx] = v;
    xn_b[swzks(idx)] = (bf16)v;
  }
}

__global__ void zero_kernel(float* p, size_t n) {
  size_t i = (size_t)blockIdx.x * blockDim.x + threadIdx.x;
  if (i < n) p[i] = 0.f;
}

__global__ __launch_bounds__(256) void stat2_partial_kernel(
    const float* __restrict__ xt, float* __restrict__ sums, int Cc, int Nn, int chunk)
{
  const int b = blockIdx.z, c = blockIdx.y * 256 + threadIdx.x;
  const int nb = blockIdx.x * chunk;
  const int ne = (nb + chunk < Nn) ? nb + chunk : Nn;
  const float* p = xt + (size_t)b * Nn * Cc + c;
  float s = 0.f, q = 0.f;
  for (int n = nb; n < ne; ++n) { float v = p[(size_t)n * Cc]; s += v; q += v * v; }
  atomicAdd(&sums[((size_t)b * Cc + c) * 2],     s);
  atomicAdd(&sums[((size_t)b * Cc + c) * 2 + 1], q);
}

__global__ void stat2_final_kernel(const float* __restrict__ sums,
                                   float* __restrict__ mean, float* __restrict__ rstd, int Nn, int total)
{
  int i = blockIdx.x * blockDim.x + threadIdx.x;
  if (i >= total) return;
  float m = sums[2 * i] / Nn;
  float v = sums[2 * i + 1] / Nn - m * m;
  mean[i] = m;
  rstd[i] = rsqrtf((v > 0.f ? v : 0.f) + 1e-5f);
}

__global__ void normT2_kernel(const float* __restrict__ xt, const float* __restrict__ mean,
                              const float* __restrict__ rstd, bf16* __restrict__ out,
                              int Cc, int Nn, size_t total)
{
  size_t i = (size_t)blockIdx.x * blockDim.x + threadIdx.x;
  if (i >= total) return;
  const int c = (int)(i % Cc);
  const int b = (int)((i / Cc) / Nn);
  const int s = b * Cc + c;
  out[swzks(i)] = (bf16)((xt[i] - mean[s]) * rstd[s]);
}

// ------------------------------- wavelets ----------------------------------
// One DWT level on t-layout data; thread = channel => fully coalesced.
__global__ __launch_bounds__(256) void dwt_kernel(
    const float* __restrict__ a, float* __restrict__ aout,
    bf16* __restrict__ aout_b, bf16* __restrict__ dout_b,
    int Na, int Ma, int Cc, Filt f)
{
  const int c = blockIdx.y * 256 + threadIdx.x;
  const int m = blockIdx.x, b = blockIdx.z;
  const float* ab = a + (size_t)b * Na * Cc;
  float sa = 0.f, sd = 0.f;
  #pragma unroll
  for (int k = 0; k < 8; ++k) {
    int e = 2 * m - 6 + k;                       // symmetric extension
    e = (e < 0) ? (-e - 1) : ((e >= Na) ? (2 * Na - 1 - e) : e);
    const float v = ab[(size_t)e * Cc + c];
    sa += f.lo[k] * v;
    sd += f.hi[k] * v;
  }
  const size_t idx  = ((size_t)b * Ma + m) * Cc + c;
  const size_t idxs = ((size_t)b * Ma + m) * Cc + swzk(c);
  aout[idx] = sa;
  if (aout_b) aout_b[idxs] = (bf16)sa;
  dout_b[idxs] = (bf16)sd;
}

// One IDWT level: out[j] = sum_k over even (j-1+k): lo[k]*cA[u] + hi[k]*cD[u]
__global__ __launch_bounds__(256) void idwt_kernel(
    const float* __restrict__ ya, const float* __restrict__ yd, float* __restrict__ out,
    int aRows, int dRows, int Mu, int Nout, int Cc, Filt f)
{
  const int c = blockIdx.y * 256 + threadIdx.x;
  const int j = blockIdx.x, b = blockIdx.z;
  const float* pa = ya + (size_t)b * aRows * Cc;
  const float* pd = yd + (size_t)b * dRows * Cc;
  float s = 0.f;
  #pragma unroll
  for (int k = 0; k < 8; ++k) {
    const int t = j - 1 + k;
    if (t >= 0 && !(t & 1)) {
      const int u = t >> 1;
      if (u < Mu)
        s += f.lo[k] * pa[(size_t)u * Cc + c] + f.hi[k] * pd[(size_t)u * Cc + c];
    }
  }
  out[((size_t)b * Nout + j) * Cc + c] = s;
}

__global__ void gelu_add_kernel(const float* __restrict__ x_t, const float* __restrict__ xw,
                                const float* __restrict__ skip, float* __restrict__ xop, size_t total)
{
  size_t i = (size_t)blockIdx.x * blockDim.x + threadIdx.x;
  if (i < total) xop[i] = x_t[i] + geluf(xw[i] + skip[i]);
}

// ------------------------------ weight prep --------------------------------
// Plain f32 -> bf16 with fragment K-swizzle (row lengths are multiples of 32).
__global__ void cvt_bf16_kernel(const float* __restrict__ s, bf16* __restrict__ d, size_t n) {
  size_t i = (size_t)blockIdx.x * blockDim.x + threadIdx.x;
  if (i < n) d[swzks(i)] = (bf16)s[i];
}
// w_w (O,C,3) -> three bf16 planes d[t][o][c], K-swizzled
__global__ void cvt_convw_kernel(const float* __restrict__ w, bf16* __restrict__ d, int Cc) {
  size_t i = (size_t)blockIdx.x * blockDim.x + threadIdx.x;
  const size_t CC = (size_t)Cc * Cc;
  if (i >= 3 * CC) return;
  const int t = (int)(i / CC);
  const size_t oc = i % CC;
  d[swzks(i)] = (bf16)w[oc * 3 + t];
}

// ---------------------------------------------------------------------------
static void build_filts(Filt* dwt, Filt* idw) {
  static const float DL[8] = {
    -0.010597401784997278f, 0.032883011666982945f, 0.030841381835986965f,
    -0.18703481171888114f, -0.02798376941698385f, 0.6308807679295904f,
     0.7148465705525415f, 0.23037781330885523f };
  float rec_lo[8], dec_hi[8];
  for (int k = 0; k < 8; ++k) rec_lo[k] = DL[7 - k];
  for (int k = 0; k < 8; ++k) dec_hi[k] = ((k & 1) ? 1.f : -1.f) * rec_lo[k];
  for (int k = 0; k < 8; ++k) { dwt->lo[k] = rec_lo[k]; dwt->hi[k] = dec_hi[7 - k]; }
  for (int k = 0; k < 8; ++k) { idw->lo[k] = DL[k];     idw->hi[k] = dec_hi[k]; }
}

extern "C" void kernel_launch(void* const* d_in, const int* in_sizes, int n_in,
                              void* d_out, int out_size, void* d_ws, size_t ws_size,
                              hipStream_t stream) {
  (void)in_sizes; (void)n_in; (void)out_size; (void)ws_size;
  constexpr int B = 8, C = 512, N = 4096, H = 1024;
  constexpr int M1 = 2051, M2 = 1029, M3 = 518, M4 = 262;

  const float* x    = (const float*)d_in[0];
  const float* w_w  = (const float*)d_in[1];
  const float* w_b  = (const float*)d_in[2];
  const float* wA_w = (const float*)d_in[3];
  const float* wA_b = (const float*)d_in[4];
  const float* wD_w = (const float*)d_in[5];
  const float* wD_b = (const float*)d_in[6];
  const float* m1_w = (const float*)d_in[7];
  const float* m1_b = (const float*)d_in[8];
  const float* m2_w = (const float*)d_in[9];
  const float* m2_b = (const float*)d_in[10];
  float* out = (float*)d_out;

  char* ws = (char*)d_ws;
  size_t off = 0;
  auto allocf = [&](size_t n) -> float* { float* p = (float*)(ws + off); off = (off + n * 4 + 15) & ~(size_t)15; return p; };
  auto allocb = [&](size_t n) -> bf16*  { bf16*  p = (bf16*)(ws + off);  off = (off + n * 2 + 15) & ~(size_t)15; return p; };

  const size_t BNC = (size_t)B * N * C;
  float* mean1 = allocf(B * C);
  float* rstd1 = allocf(B * C);
  float* x_t   = allocf(BNC);
  float* xn_t  = allocf(BNC);
  float* skip  = allocf(BNC);
  float* a1    = allocf((size_t)B * M1 * C);
  float* a2    = allocf((size_t)B * M2 * C);
  float* a3    = allocf((size_t)B * M3 * C);
  float* a4    = allocf((size_t)B * M4 * C);
  float* pwA   = allocf((size_t)B * M4 * C);
  float* pwD4  = allocf((size_t)B * M4 * C);
  float* pwD3  = allocf((size_t)B * M3 * C);
  float* pwD2  = allocf((size_t)B * M2 * C);
  float* pwD1  = allocf((size_t)B * M1 * C);
  float* r3    = allocf((size_t)B * 518  * C);
  float* r2    = allocf((size_t)B * 1030 * C);
  float* r1    = allocf((size_t)B * 2052 * C);
  float* xw    = allocf(BNC);
  float* xop   = allocf(BNC);
  float* sums2 = allocf((size_t)B * C * 2);
  float* mean2 = allocf(B * C);
  float* rstd2 = allocf(B * C);
  bf16* zrow   = allocb(1152);          // zero row for OOB GEMM reads (K<=1024)
  bf16* xn_b   = allocb(BNC);
  bf16* a4_b   = allocb((size_t)B * M4 * C);
  bf16* d1_b   = allocb((size_t)B * M1 * C);
  bf16* d2_b   = allocb((size_t)B * M2 * C);
  bf16* d3_b   = allocb((size_t)B * M3 * C);
  bf16* d4_b   = allocb((size_t)B * M4 * C);
  bf16* xopn_b = allocb(BNC);
  bf16* h_b    = allocb((size_t)B * N * H);
  bf16* wc_b   = allocb((size_t)3 * C * C);
  bf16* wA_wb  = allocb((size_t)C * C);
  bf16* wD_wb  = allocb((size_t)4 * C * C);
  bf16* m1_wb  = allocb((size_t)H * C);
  bf16* m2_wb  = allocb((size_t)C * H);

  Filt fd, fi;
  build_filts(&fd, &fi);

  auto cdiv = [](size_t a, size_t b) { return (unsigned)((a + b - 1) / b); };

  // --- weights -> bf16 (K-swizzled), zero row --------------------------
  zero_kernel<<<cdiv(576, 256), 256, 0, stream>>>((float*)zrow, 576);
  cvt_convw_kernel<<<cdiv((size_t)3 * C * C, 256), 256, 0, stream>>>(w_w, wc_b, C);
  cvt_bf16_kernel<<<cdiv((size_t)C * C, 256), 256, 0, stream>>>(wA_w, wA_wb, (size_t)C * C);
  cvt_bf16_kernel<<<cdiv((size_t)4 * C * C, 256), 256, 0, stream>>>(wD_w, wD_wb, (size_t)4 * C * C);
  cvt_bf16_kernel<<<cdiv((size_t)H * C, 256), 256, 0, stream>>>(m1_w, m1_wb, (size_t)H * C);
  cvt_bf16_kernel<<<cdiv((size_t)C * H, 256), 256, 0, stream>>>(m2_w, m2_wb, (size_t)C * H);

  // --- instance norm + transpose to (B,N,C) ----------------------------
  rowstat_kernel<<<B * C, 256, 0, stream>>>(x, mean1, rstd1, N);
  normT_kernel<<<dim3(N / 32, C / 32, B), dim3(32, 8), 0, stream>>>(
      x, mean1, rstd1, x_t, xn_t, xn_b, C, N);

  // --- conv3 skip = 3 shifted WMMA GEMM passes -------------------------
  for (int t = 0; t < 3; ++t)
    wmma_gemm_kernel<<<dim3(N / 128, C / 64, B), 256, 0, stream>>>(
        wc_b + (size_t)t * C * C, xn_b, zrow, w_b, skip, nullptr, nullptr,
        C, C, N, t - 1, 0, t > 0);

  // --- 4-level DWT -----------------------------------------------------
  dwt_kernel<<<dim3(M1, C / 256, B), 256, 0, stream>>>(xn_t, a1, nullptr, d1_b, N,  M1, C, fd);
  dwt_kernel<<<dim3(M2, C / 256, B), 256, 0, stream>>>(a1,   a2, nullptr, d2_b, M1, M2, C, fd);
  dwt_kernel<<<dim3(M3, C / 256, B), 256, 0, stream>>>(a2,   a3, nullptr, d3_b, M2, M3, C, fd);
  dwt_kernel<<<dim3(M4, C / 256, B), 256, 0, stream>>>(a3,   a4, a4_b,    d4_b, M3, M4, C, fd);

  // --- per-scale pointwise GEMMs --------------------------------------
  wmma_gemm_kernel<<<dim3(cdiv(M4, 128), C / 64, B), 256, 0, stream>>>(
      wA_wb, a4_b, zrow, wA_b, pwA, nullptr, nullptr, C, C, M4, 0, 0, 0);
  wmma_gemm_kernel<<<dim3(cdiv(M4, 128), C / 64, B), 256, 0, stream>>>(
      wD_wb + 0 * (size_t)C * C, d4_b, zrow, wD_b + 0 * C, pwD4, nullptr, nullptr, C, C, M4, 0, 0, 0);
  wmma_gemm_kernel<<<dim3(cdiv(M3, 128), C / 64, B), 256, 0, stream>>>(
      wD_wb + 1 * (size_t)C * C, d3_b, zrow, wD_b + 1 * C, pwD3, nullptr, nullptr, C, C, M3, 0, 0, 0);
  wmma_gemm_kernel<<<dim3(cdiv(M2, 128), C / 64, B), 256, 0, stream>>>(
      wD_wb + 2 * (size_t)C * C, d2_b, zrow, wD_b + 2 * C, pwD2, nullptr, nullptr, C, C, M2, 0, 0, 0);
  wmma_gemm_kernel<<<dim3(cdiv(M1, 128), C / 64, B), 256, 0, stream>>>(
      wD_wb + 3 * (size_t)C * C, d1_b, zrow, wD_b + 3 * C, pwD1, nullptr, nullptr, C, C, M1, 0, 0, 0);

  // --- waverec (with trims) -------------------------------------------
  idwt_kernel<<<dim3(518,  C / 256, B), 256, 0, stream>>>(pwA, pwD4, r3, M4,  M4, M4,  518,  C, fi);
  idwt_kernel<<<dim3(1030, C / 256, B), 256, 0, stream>>>(r3,  pwD3, r2, 518, M3, 518, 1030, C, fi);
  idwt_kernel<<<dim3(2052, C / 256, B), 256, 0, stream>>>(r2,  pwD2, r1, 1030, M2, M2, 2052, C, fi);
  idwt_kernel<<<dim3(N,    C / 256, B), 256, 0, stream>>>(r1,  pwD1, xw, 2052, M1, M1, N,    C, fi);

  // --- x_op = x + gelu(xw + skip) -------------------------------------
  gelu_add_kernel<<<cdiv(BNC, 256), 256, 0, stream>>>(x_t, xw, skip, xop, BNC);

  // --- inorm(x_op) -> bf16 --------------------------------------------
  zero_kernel<<<cdiv((size_t)B * C * 2, 256), 256, 0, stream>>>(sums2, (size_t)B * C * 2);
  stat2_partial_kernel<<<dim3(N / 128, C / 256, B), 256, 0, stream>>>(xop, sums2, C, N, 128);
  stat2_final_kernel<<<cdiv((size_t)B * C, 256), 256, 0, stream>>>(sums2, mean2, rstd2, N, B * C);
  normT2_kernel<<<cdiv(BNC, 256), 256, 0, stream>>>(xop, mean2, rstd2, xopn_b, C, N, BNC);

  // --- MLP: h = gelu(m1 @ xn); out = x_op + m2 @ h + b2 ----------------
  wmma_gemm_kernel<<<dim3(N / 128, H / 64, B), 256, 0, stream>>>(
      m1_wb, xopn_b, zrow, m1_b, nullptr, h_b, nullptr, H, C, N, 0, 2, 0);
  wmma_gemm_kernel<<<dim3(N / 128, C / 64, B), 256, 0, stream>>>(
      m2_wb, h_b, zrow, m2_b, out, nullptr, xop, C, H, N, 0, 3, 0);
}
